// Cif_74775380623400
// MI455X (gfx1250) — matrix operand order repaired and checked by reading, not compiled, not run on hardware
//
#include <hip/hip_runtime.h>
#include <cstdint>
#include <cstddef>

// ---------------- problem constants (from reference) ----------------
#define B_  32
#define T_  3000
#define TP_ 3012                 // halo-padded rows per batch: [0,1]=0, [2..3001]=t, rest 0
#define C_  512
#define K5_ 5
#define KC_ 2560                 // 5*C  (im2col K)
#define TILES_T_ 188             // ceil(T/16)
#define CS_ELEMS 49152000LL      // B*T*C
#define HSB_ELEMS 49348608LL     // B*TP*C
#define BT_ 96000                // B*T

typedef __bf16 bf16_t;
typedef bf16_t v16bf __attribute__((ext_vector_type(16)));
typedef float  v8f   __attribute__((ext_vector_type(8)));

union Frag32B { v16bf v; uint4 q[2]; };

__device__ __forceinline__ void atomAddF(float* p, float v) {
  unsafeAtomicAdd(p, v);       // global_atomic_add_f32 / ds_add_f32
}

// ---------------- prep: conv_w [O, Cin, 5] f32 -> wB [O, 5, Cin] bf16 ---------------
__global__ void cvt_w_kernel(const float* __restrict__ conv_w, bf16_t* __restrict__ wB) {
  int idx = blockIdx.x * blockDim.x + threadIdx.x;            // dst index
  if (idx >= C_ * KC_) return;
  int o   = idx / KC_;
  int r   = idx - o * KC_;
  int k   = r >> 9;              // r / 512
  int cin = r & 511;
  wB[idx] = (bf16_t)conv_w[(size_t)o * KC_ + (size_t)cin * K5_ + k];
}

// ---------------- prep: hs_pad f32 -> halo-padded bf16 [B][TP][C] -------------------
// padded row pr maps to t = pr - 2; rows outside [0,T) are zero (conv zero padding AND
// ragged-tile padding), so the GEMM A-loads are always unconditional and in-bounds.
__global__ void cvt_hs_kernel(const float* __restrict__ hs, bf16_t* __restrict__ hsB) {
  long long e = (long long)(blockIdx.x * (long long)blockDim.x + threadIdx.x) * 4;
  if (e >= HSB_ELEMS) return;
  int b = (int)(e / ((long long)TP_ * C_));
  long long within = e - (long long)b * TP_ * C_;
  int pr = (int)(within >> 9);                   // padded row
  int c  = (int)(within & 511);
  int t  = pr - 2;
  union { bf16_t h[4]; uint2 u; } p;
  if (t >= 0 && t < T_) {
    float4 f = *(const float4*)(hs + ((size_t)b * T_ + t) * C_ + c);
    p.h[0] = (bf16_t)f.x; p.h[1] = (bf16_t)f.y;
    p.h[2] = (bf16_t)f.z; p.h[3] = (bf16_t)f.w;
  } else {
    p.u = make_uint2(0u, 0u);
  }
  *(uint2*)(hsB + e) = p.u;
}

// ---------------- zero the cs region of d_out + the loss accumulator ----------------
__global__ void zero_kernel(float* __restrict__ cs, float* __restrict__ loss_acc) {
  long long i4 = (long long)(blockIdx.x * (long long)blockDim.x + threadIdx.x) * 4;
  if (i4 < CS_ELEMS) *(float4*)(cs + i4) = make_float4(0.f, 0.f, 0.f, 0.f);
  if (i4 == 0) *loss_acc = 0.f;
}

// ---------------- fused conv(k=5) GEMM + relu + linear + sigmoid -> alpha -----------
// Block = 512 threads (16 waves). Block owns a 16-row t-tile of one batch and the
// FULL N=512 output columns (each wave: two 16x16 N-tiles).
// K loop: 5 taps x 16 cin-steps of 32 -> all loads are immediate-offset global_b128.
__global__ __launch_bounds__(512) void conv_alpha_kernel(
    const bf16_t* __restrict__ hsB, const bf16_t* __restrict__ wB,
    const float* __restrict__ conv_b, const float* __restrict__ lin_w,
    const float* __restrict__ lin_b, const int* __restrict__ hs_mask,
    float* __restrict__ alpha_ws, float* __restrict__ loss_acc)
{
  const int b    = blockIdx.x / TILES_T_;
  const int m0   = (blockIdx.x % TILES_T_) * 16;
  const int tid  = threadIdx.x;
  const int wave = tid >> 5;
  const int lane = tid & 31;
  const int row  = lane & 15;
  const int khalf = lane >> 4;

  __shared__ float s_part[16][16];               // [wave][M]

  const int tl = m0 + row;                       // A-fragment row (local t)
  // padded A row for tap k is (tl + k); halo guarantees validity for tl<3008, k<5
  const bf16_t* arow0 = hsB + ((size_t)b * TP_ + tl) * C_ + khalf * 16;

  const int ob0 = wave * 32;                     // N-tile 0 column base
  const int ob1 = ob0 + 16;                      // N-tile 1 column base
  const bf16_t* wrow0 = wB + (size_t)(ob0 + row) * KC_ + khalf * 16;
  const bf16_t* wrow1 = wB + (size_t)(ob1 + row) * KC_ + khalf * 16;

  v8f acc0 = {};
  v8f acc1 = {};

  for (int k = 0; k < K5_; ++k) {
    const uint4* pa  = (const uint4*)(arow0 + (size_t)k * C_);
    const uint4* pb0 = (const uint4*)(wrow0 + (size_t)k * C_);
    const uint4* pb1 = (const uint4*)(wrow1 + (size_t)k * C_);
    #pragma unroll
    for (int ci = 0; ci < 16; ++ci) {            // 16 steps of 32 cin
      // A: 16x32 bf16; lane holds K run [16*khalf..+15]
      Frag32B a;
      a.q[0] = pa[ci * 4];
      a.q[1] = pa[ci * 4 + 1];                   // +8 elements
      // B: 32x16 bf16; lane holds column o, K run [16*khalf..+15]
      Frag32B f0, f1;
      f0.q[0] = pb0[ci * 4]; f0.q[1] = pb0[ci * 4 + 1];
      f1.q[0] = pb1[ci * 4]; f1.q[1] = pb1[ci * 4 + 1];

      acc0 = __builtin_amdgcn_wmma_f32_16x16x32_bf16(false, a.v, false, f0.v,
                                                     (short)0, acc0, false, false);
      acc1 = __builtin_amdgcn_wmma_f32_16x16x32_bf16(false, a.v, false, f1.v,
                                                     (short)0, acc1, false, false);
    }
  }

  // relu + bias + dot with lin_w, reduced over o.
  // C/D layout: lane holds column o = base + (lane&15), rows M = 8*khalf + g.
  const int o0 = ob0 + row, o1 = ob1 + row;
  const float bias0 = conv_b[o0], lw0 = lin_w[o0];
  const float bias1 = conv_b[o1], lw1 = lin_w[o1];

  #pragma unroll
  for (int g = 0; g < 8; ++g) {
    float x = fmaxf(acc0[g] + bias0, 0.f) * lw0
            + fmaxf(acc1[g] + bias1, 0.f) * lw1;
    // butterfly over the 16 lanes of this half (same M across them)
    x += __shfl_xor(x, 1, 32);
    x += __shfl_xor(x, 2, 32);
    x += __shfl_xor(x, 4, 32);
    x += __shfl_xor(x, 8, 32);
    if (row == 0) s_part[wave][khalf * 8 + g] = x;
  }
  __syncthreads();

  if (tid < 16) {
    float sum = 0.f;
    #pragma unroll
    for (int w = 0; w < 16; ++w) sum += s_part[w][tid];
    const int t = m0 + tid;
    if (t < T_) {
      const float apre = 1.f / (1.f + __expf(-(sum + lin_b[0])));
      const int  mk = hs_mask[b * T_ + t];       // [B,1,T]
      alpha_ws[b * T_ + t] = (mk != 0) ? apre : 0.f;
      atomAddF(loss_acc, apre);                  // loss uses pre-mask alpha
    }
  }
}

// ---------------- per-batch inclusive prefix sum of alpha + fire counts -------------
__global__ __launch_bounds__(256) void scan_kernel(const float* __restrict__ alpha,
                                                   float* __restrict__ S,
                                                   int* __restrict__ nvalid)
{
  const int b = blockIdx.x;
  const int tid = threadIdx.x;
  const int base = b * T_;
  const int CH = 12;                              // 256*12 >= 3000
  __shared__ float part[256];

  float local = 0.f;
  const int t0 = tid * CH;
  for (int j = 0; j < CH; ++j) {
    int t = t0 + j;
    if (t < T_) local += alpha[base + t];
  }
  part[tid] = local;
  __syncthreads();
  for (int off = 1; off < 256; off <<= 1) {
    float add = 0.f;
    if (tid >= off) add = part[tid - off];
    __syncthreads();
    part[tid] += add;
    __syncthreads();
  }
  float running = part[tid] - local;              // exclusive prefix of chunks
  for (int j = 0; j < CH; ++j) {
    int t = t0 + j;
    if (t < T_) { running += alpha[base + t]; S[base + t] = running; }
  }
  if (tid == 0) {
    float ST = part[255];
    int n = (int)floorf(ST);
    nvalid[b] = n + ((ST - (float)n) >= 0.5f ? 1 : 0);   // tail emission
  }
}

// ---------------- parallel CIF: segmented weighted scatter cs[b,r] += w*hs[b,t] -----
__global__ __launch_bounds__(256) void scatter_kernel(
    const float* __restrict__ hs, const float* __restrict__ alpha,
    const float* __restrict__ S, const int* __restrict__ nvalid,
    float* __restrict__ cs)
{
  const int bt = blockIdx.x;                      // one (b,t) row per block
  const int b = bt / T_;
  const int t = bt - b * T_;

  const float Sc = S[bt];
  const float Sp = (t == 0) ? 0.f : S[bt - 1];
  const int np = (int)floorf(Sp);
  const int nc = (int)floorf(Sc);
  const bool fired = nc > np;
  const float a = alpha[bt];
  const int nv = nvalid[b];

  const float w1 = fired ? (1.f - (Sp - (float)np)) : a;  // into row np
  const float w2 = Sc - (float)nc;                        // leftover into row nc
  const bool c1 = (np < nv) && (w1 != 0.f);
  const bool c2 = fired && (nc < nv) && (w2 != 0.f);
  if (!c1 && !c2) return;

  const float* hrow = hs + (size_t)bt * C_;
  float* o1 = cs + ((size_t)b * T_ + np) * C_;
  float* o2 = cs + ((size_t)b * T_ + nc) * C_;

  for (int c = threadIdx.x; c < C_; c += 256) {
    float h = hrow[c];
    if (c1) atomAddF(o1 + c, w1 * h);
    if (c2) atomAddF(o2 + c, w2 * h);
  }
}

// ---------------- cs_mask (rows < nvalid are nonzero emissions) + loss --------------
__global__ void mask_loss_kernel(const int* __restrict__ nvalid,
                                 const float* __restrict__ loss_acc,
                                 float* __restrict__ mask_out,
                                 float* __restrict__ loss_out)
{
  int idx = blockIdx.x * blockDim.x + threadIdx.x;
  if (idx >= BT_) return;
  int b = idx / T_;
  int t = idx - b * T_;
  mask_out[idx] = (t < nvalid[b]) ? 1.f : 0.f;
  if (idx == 0) *loss_out = *loss_acc;            // alpha>0 -> abs is identity
}

// ---------------- host-side orchestration -------------------------------------------
extern "C" void kernel_launch(void* const* d_in, const int* in_sizes, int n_in,
                              void* d_out, int out_size, void* d_ws, size_t ws_size,
                              hipStream_t stream) {
  const float* hs_pad  = (const float*)d_in[0];
  const int*   hs_mask = (const int*)  d_in[1];
  const float* conv_w  = (const float*)d_in[2];
  const float* conv_b  = (const float*)d_in[3];
  const float* lin_w   = (const float*)d_in[4];
  const float* lin_b   = (const float*)d_in[5];

  float* cs_out   = (float*)d_out;                         // [B,T,C]
  float* mask_out = (float*)d_out + CS_ELEMS;              // [B,1,T]
  float* loss_out = (float*)d_out + CS_ELEMS + BT_;        // scalar

  // workspace carve-up (bytes)
  char* ws = (char*)d_ws;
  bf16_t* hsB      = (bf16_t*)(ws);                        //  98,697,216 B
  bf16_t* wB       = (bf16_t*)(ws + 98697216);             //   2,621,440 B
  float*  alpha_ws = (float*) (ws + 101318656);            //     384,000 B
  float*  S_ws     = (float*) (ws + 101702656);            //     384,000 B
  int*    nvalid   = (int*)   (ws + 102086656);            //         128 B
  float*  loss_acc = (float*) (ws + 102086784);            //           4 B

  // 1. zero cs + loss accumulator
  zero_kernel<<<(int)(CS_ELEMS / 4 / 256), 256, 0, stream>>>(cs_out, loss_acc);
  // 2. weight transpose+convert [O,Cin,5] -> [O,5,Cin] bf16
  cvt_w_kernel<<<(C_ * KC_ + 255) / 256, 256, 0, stream>>>(conv_w, wB);
  // 3. activations f32 -> halo-padded bf16
  cvt_hs_kernel<<<(int)(HSB_ELEMS / 4 / 256), 256, 0, stream>>>(hs_pad, hsB);
  // 4. fused conv-GEMM (WMMA bf16) + relu + linear + sigmoid -> alpha
  conv_alpha_kernel<<<B_ * TILES_T_, 512, 0, stream>>>(
      hsB, wB, conv_b, lin_w, lin_b, hs_mask, alpha_ws, loss_acc);
  // 5. per-batch prefix sums of alpha (fire schedule)
  scan_kernel<<<B_, 256, 0, stream>>>(alpha_ws, S_ws, nvalid);
  // 6. parallel integrate-and-fire via weighted scatter
  scatter_kernel<<<BT_, 256, 0, stream>>>(hs_pad, alpha_ws, S_ws, nvalid, cs_out);
  // 7. cs_mask + loss
  mask_loss_kernel<<<(BT_ + 255) / 256, 256, 0, stream>>>(nvalid, loss_acc,
                                                          mask_out, loss_out);
}
